// CrossAttention_90366111908144
// MI455X (gfx1250) — compile-verified, matrix operand order checked by reading, not run on hardware
//
#include <hip/hip_runtime.h>
#include <hip/hip_bf16.h>
#include <math.h>

typedef __attribute__((ext_vector_type(16))) _Float16 v16h;
typedef __attribute__((ext_vector_type(8)))  float    v8f;

#define T_ 4
#define N_ 16
#define C_ 256
#define V_ 1024
#define NBATCH (T_ * N_)          // 64 (t,n) batches
#define KT_ 8                     // 256 / 32
#define VT_ 64                    // 1024 / 16
#define MT_ 16                    // 256 / 16
#define FRAG_HALVES 512           // 32x16 fp16 fragment, lane-ordered
#define BPACK_BATCH (KT_ * VT_ * FRAG_HALVES)   // 262144 halves per (t,n)

// k index inside a 16x32 (or 32x16) fp16 fragment: half-slot i (0..15), lane-half hi (0/1)
__device__ __forceinline__ int frag_k(int i, int hi) {
    return (i < 8) ? (hi * 8 + i) : (8 + hi * 8 + i);
}
// inverse: given kr in [0,32) -> (hi, i)
__device__ __forceinline__ void frag_k_inv(int kr, int& hi, int& i) {
    hi = (kr >> 3) & 1;
    i  = (kr & 7) | (((kr >> 4) & 1) << 3);
}

// ---------------- pack a 256x256 f32 weight matrix into WMMA A-fragments ----------------
__global__ void pack_weights(const float* __restrict__ W, _Float16* __restrict__ P) {
    int idx   = blockIdx.x * 256 + threadIdx.x;      // 65536 packed halves
    int i     = idx & 15;
    int lane  = (idx >> 4) & 31;
    int frag  = idx >> 9;                            // mt*8 + kt
    int kt    = frag & 7, mt = frag >> 3;
    int m     = lane & 15, hi = lane >> 4;
    int k     = frag_k(i, hi);
    P[idx] = (_Float16)W[(mt * 16 + m) * C_ + (kt * 32 + k)];
}

// ---------------- IF-node over T + pack spikes into WMMA B-fragments ----------------
// in: (N,T,C,V) f32.  out: per batch (t*N+n): [kt][vt][lane][i] fp16 spikes.
__global__ void spike_pack(const float* __restrict__ X, _Float16* __restrict__ P, float th) {
    int idx = blockIdx.x * 256 + threadIdx.x;        // N*C*V = 4M threads
    int v = idx & (V_ - 1);
    int c = (idx >> 10) & (C_ - 1);
    int n = idx >> 18;
    int kt = c >> 5, kr = c & 31;
    int vt = v >> 4, vr = v & 15;
    int hi, i; frag_k_inv(kr, hi, i);
    size_t poff = (size_t)(kt * VT_ + vt) * FRAG_HALVES + (size_t)(vr + (hi << 4)) * 16 + i;
    float mem = 0.f;
    #pragma unroll
    for (int t = 0; t < T_; ++t) {
        mem += X[(((size_t)n * T_ + t) * C_ + c) * V_ + v];
        float s = (mem - th >= 0.f) ? 1.f : 0.f;
        mem *= (1.f - s);
        P[(size_t)(t * N_ + n) * BPACK_BATCH + poff] = (_Float16)s;
    }
}

// ---------------- WMMA GEMM: C[batch][256][1024] = A(256x256) x B(256x1024) ----------------
// 8 waves/block; all 8 waves of a block share the SAME B fragments (same vg) and cover
// all 8 m-groups -> B reads hit WGP$ instead of L2. 64 v_wmma per wave.
__global__ void __launch_bounds__(256)
gemm_wmma(const _Float16* __restrict__ Apack, const _Float16* __restrict__ Bpack,
          float* __restrict__ Cout) {
    int wave = threadIdx.x >> 5;
    int lane = threadIdx.x & 31;
    int job  = blockIdx.x * 8 + wave;                // NBATCH * 128 jobs
    int batch = job >> 7;
    int r     = job & 127;
    int mg    = r & 7;                               // waves in a block: mg = 0..7
    int vg    = r >> 3;                              // shared across the block's 8 waves
    const _Float16* Bb = Bpack + (size_t)batch * BPACK_BATCH
                               + (size_t)(vg * 4) * FRAG_HALVES + (size_t)lane * 16;
    const _Float16* Ab = Apack + (size_t)(mg * 2 * KT_) * FRAG_HALVES + (size_t)lane * 16;

    v8f acc[2][4];
    #pragma unroll
    for (int a = 0; a < 2; ++a)
        #pragma unroll
        for (int b = 0; b < 4; ++b) acc[a][b] = (v8f){0,0,0,0,0,0,0,0};

    for (int kt = 0; kt < KT_; ++kt) {
        v16h a0 = *(const v16h*)(Ab + (size_t)kt * FRAG_HALVES);
        v16h a1 = *(const v16h*)(Ab + (size_t)(KT_ + kt) * FRAG_HALVES);
        #pragma unroll
        for (int j = 0; j < 4; ++j) {
            v16h b = *(const v16h*)(Bb + (size_t)(kt * VT_ + j) * FRAG_HALVES);
            acc[0][j] = __builtin_amdgcn_wmma_f32_16x16x32_f16(false, a0, false, b,
                            (short)0, acc[0][j], false, false);
            acc[1][j] = __builtin_amdgcn_wmma_f32_16x16x32_f16(false, a1, false, b,
                            (short)0, acc[1][j], false, false);
        }
    }

    int hi = lane >> 4, nl = lane & 15;
    float* Cb = Cout + (size_t)batch * C_ * V_;
    #pragma unroll
    for (int i2 = 0; i2 < 2; ++i2)
        #pragma unroll
        for (int j = 0; j < 4; ++j) {
            int v = (vg*4 + j) * 16 + nl;
            #pragma unroll
            for (int rr = 0; rr < 8; ++rr) {
                int c = (mg*2 + i2) * 16 + rr + 8*hi;
                Cb[(size_t)c * V_ + v] = acc[i2][j][rr];
            }
        }
}

// ---------------- BN + IF on k,v conv; kv = sum_v k_s*v_s; IF(0.5) over T -> kv_s ----------------
__global__ void kv_reduce(const float* __restrict__ kconv, const float* __restrict__ vconv,
                          const float* kg, const float* kb, const float* km, const float* kvar,
                          const float* vg, const float* vb, const float* vm, const float* vvar,
                          float* __restrict__ kv_s) {
    int n = blockIdx.x >> 8;
    int c = blockIdx.x & 255;
    int tid = threadIdx.x;
    float ksc = kg[c] * rsqrtf(kvar[c] + 1e-5f), ksh = kb[c] - km[c] * ksc;
    float vsc = vg[c] * rsqrtf(vvar[c] + 1e-5f), vsh = vb[c] - vm[c] * vsc;
    float kmem[4] = {0,0,0,0}, vmem[4] = {0,0,0,0};
    __shared__ float red[256];
    __shared__ float kvt[T_];
    for (int t = 0; t < T_; ++t) {
        const float* kp = kconv + ((size_t)(t*N_ + n) * C_ + c) * V_;
        const float* vp = vconv + ((size_t)(t*N_ + n) * C_ + c) * V_;
        float p = 0.f;
        #pragma unroll
        for (int j = 0; j < 4; ++j) {
            int v = tid + 256 * j;
            kmem[j] += kp[v] * ksc + ksh;
            float ks = (kmem[j] - 1.f >= 0.f) ? 1.f : 0.f;
            kmem[j] *= (1.f - ks);
            vmem[j] += vp[v] * vsc + vsh;
            float vs = (vmem[j] - 1.f >= 0.f) ? 1.f : 0.f;
            vmem[j] *= (1.f - vs);
            p += ks * vs;
        }
        red[tid] = p; __syncthreads();
        for (int s = 128; s > 0; s >>= 1) {
            if (tid < s) red[tid] += red[tid + s];
            __syncthreads();
        }
        if (tid == 0) kvt[t] = red[0];
        __syncthreads();
    }
    if (tid == 0) {
        float mem = 0.f;
        #pragma unroll
        for (int t = 0; t < T_; ++t) {
            mem += kvt[t];
            float s = (mem - 0.5f >= 0.f) ? 1.f : 0.f;
            mem *= (1.f - s);
            kv_s[(size_t)(t*N_ + n) * C_ + c] = s;
        }
    }
}

// ---------------- BN + IF on q conv, multiply by kv_s broadcast, pack for proj GEMM ----------------
__global__ void q_mul_pack(const float* __restrict__ qconv, const float* __restrict__ kv_s,
                           const float* qg, const float* qb, const float* qm, const float* qvar,
                           _Float16* __restrict__ P) {
    int idx = blockIdx.x * 256 + threadIdx.x;
    int v = idx & (V_ - 1);
    int c = (idx >> 10) & (C_ - 1);
    int n = idx >> 18;
    float sc = qg[c] * rsqrtf(qvar[c] + 1e-5f), sh = qb[c] - qm[c] * sc;
    int kt = c >> 5, kr = c & 31;
    int vt = v >> 4, vr = v & 15;
    int hi, i; frag_k_inv(kr, hi, i);
    size_t poff = (size_t)(kt * VT_ + vt) * FRAG_HALVES + (size_t)(vr + (hi << 4)) * 16 + i;
    float mem = 0.f;
    #pragma unroll
    for (int t = 0; t < T_; ++t) {
        int b = t * N_ + n;
        mem += qconv[((size_t)b * C_ + c) * V_ + v] * sc + sh;
        float s = (mem - 1.f >= 0.f) ? 1.f : 0.f;
        mem *= (1.f - s);
        float o = s * kv_s[(size_t)b * C_ + c];
        P[(size_t)b * BPACK_BATCH + poff] = (_Float16)o;
    }
}

// ---------------- bias(folded) + BN + LIF(tau=2, th=1) -> output (N,T,C,V) ----------------
__global__ void final_lif(const float* __restrict__ pconv, const float* __restrict__ pbias,
                          const float* pg, const float* pb, const float* pm, const float* pvar,
                          float* __restrict__ out) {
    int idx = blockIdx.x * 256 + threadIdx.x;
    int v = idx & (V_ - 1);
    int c = (idx >> 10) & (C_ - 1);
    int n = idx >> 18;
    float sc = pg[c] * rsqrtf(pvar[c] + 1e-5f);
    float sh = pb[c] - pm[c] * sc + pbias[c] * sc;   // fold conv bias into BN shift
    float mem = 0.f;
    #pragma unroll
    for (int t = 0; t < T_; ++t) {
        float val = pconv[((size_t)(t*N_ + n) * C_ + c) * V_ + v] * sc + sh;
        mem = 0.5f * (mem + val);                    // v += (x - v)/tau, tau = 2
        float s = (mem - 1.f >= 0.f) ? 1.f : 0.f;
        mem *= (1.f - s);
        out[(((size_t)n * T_ + t) * C_ + c) * V_ + v] = s;
    }
}

extern "C" void kernel_launch(void* const* d_in, const int* in_sizes, int n_in,
                              void* d_out, int out_size, void* d_ws, size_t ws_size,
                              hipStream_t stream) {
    const float* x      = (const float*)d_in[0];
    const float* y      = (const float*)d_in[1];
    const float* q_w    = (const float*)d_in[2];
    const float* k_w    = (const float*)d_in[3];
    const float* v_w    = (const float*)d_in[4];
    const float* proj_w = (const float*)d_in[5];
    const float* proj_b = (const float*)d_in[6];
    const float* qg = (const float*)d_in[7],  *qb = (const float*)d_in[8];
    const float* qm = (const float*)d_in[9],  *qv = (const float*)d_in[10];
    const float* kg = (const float*)d_in[11], *kb = (const float*)d_in[12];
    const float* km = (const float*)d_in[13], *kv = (const float*)d_in[14];
    const float* vg = (const float*)d_in[15], *vb = (const float*)d_in[16];
    const float* vm = (const float*)d_in[17], *vv = (const float*)d_in[18];
    const float* pg = (const float*)d_in[19], *pb = (const float*)d_in[20];
    const float* pm = (const float*)d_in[21], *pv = (const float*)d_in[22];
    float* out = (float*)d_out;

    char* ws = (char*)d_ws;
    const size_t WSZ   = 65536ull * sizeof(_Float16);          // 128 KB per packed weight
    const size_t SPKSZ = (size_t)NBATCH * BPACK_BATCH * 2;     // 32 MB per packed spike tensor
    const size_t CONV  = (size_t)NBATCH * C_ * V_ * 4;         // 64 MB per conv output
    _Float16* wq   = (_Float16*)(ws);
    _Float16* wk   = (_Float16*)(ws + WSZ);
    _Float16* wv   = (_Float16*)(ws + 2*WSZ);
    _Float16* wp   = (_Float16*)(ws + 3*WSZ);
    _Float16* xs   = (_Float16*)(ws + 4*WSZ);
    _Float16* ys   = (_Float16*)(ws + 4*WSZ + SPKSZ);
    float*    qcv  = (float*)   (ws + 4*WSZ + 2*SPKSZ);
    float*    kcv  = (float*)   (ws + 4*WSZ + 2*SPKSZ + CONV);
    float*    vcv  = (float*)   (ws + 4*WSZ + 2*SPKSZ + 2*CONV);
    float*    kvs  = (float*)   (ws + 4*WSZ + 2*SPKSZ + 3*CONV);
    _Float16* attn = xs;   // xs fully consumed before attn is produced
    float*    pcv  = qcv;  // qconv fully consumed before proj conv is produced

    // 1. pack weights
    pack_weights<<<256, 256, 0, stream>>>(q_w,    wq);
    pack_weights<<<256, 256, 0, stream>>>(k_w,    wk);
    pack_weights<<<256, 256, 0, stream>>>(v_w,    wv);
    pack_weights<<<256, 256, 0, stream>>>(proj_w, wp);

    // 2. IF spikes on x, y (th = 0.5) + fragment packing
    spike_pack<<<16384, 256, 0, stream>>>(x, xs, 0.5f);
    spike_pack<<<16384, 256, 0, stream>>>(y, ys, 0.5f);

    // 3. q/k/v 1x1-conv GEMMs (WMMA)
    gemm_wmma<<<1024, 256, 0, stream>>>(wq, xs, qcv);
    gemm_wmma<<<1024, 256, 0, stream>>>(wk, xs, kcv);
    gemm_wmma<<<1024, 256, 0, stream>>>(wv, ys, vcv);

    // 4. kv per-channel reduction + IF(0.5)
    kv_reduce<<<N_ * C_, 256, 0, stream>>>(kcv, vcv, kg, kb, km, kv, vg, vb, vm, vv, kvs);

    // 5. q spike * kv_s, packed for proj GEMM
    q_mul_pack<<<16384, 256, 0, stream>>>(qcv, kvs, qg, qb, qm, qv, attn);

    // 6. proj GEMM (WMMA)
    gemm_wmma<<<1024, 256, 0, stream>>>(wp, attn, pcv);

    // 7. bias + BN + LIF -> (N,T,C,V)
    final_lif<<<16384, 256, 0, stream>>>(pcv, proj_b, pg, pb, pm, pv, out);
}
